// ChunkMSARowAttentionWithPairBias_858993459586
// MI455X (gfx1250) — matrix operand order repaired and checked by reading, not compiled, not run on hardware
//
#include <hip/hip_runtime.h>
#include <hip/hip_bf16.h>
#include <math.h>

// ---------------------------------------------------------------------------
// ChunkMSARowAttentionWithPairBias for MI455X (gfx1250), wave32 + WMMA bf16
// + async global->LDS staging (ASYNCcnt) per CDNA5 ISA.
// ---------------------------------------------------------------------------

typedef __bf16 bf16_t;
typedef bf16_t v16bf __attribute__((ext_vector_type(16)));
typedef bf16_t v8bf  __attribute__((ext_vector_type(8)));
typedef float  v8f   __attribute__((ext_vector_type(8)));

#define N_SEQ  128
#define N_RES  256
#define DNODE  256
#define DPAIR  128
#define NHEAD  8
#define CDIM   32
#define EDIM   1024           // fused: q(256) k(256) v(256) gate(256)
#define OFF_K  256
#define OFF_V  512
#define OFF_G  768
#define QK_SCALE 0.17677669529663687f  // 1/sqrt(32)

// workspace layout (bytes)
#define WS_PB     ((size_t)0)                       // pair_bias f32 [8][256][256] = 2 MB
#define WS_MLN    ((size_t)2097152)                 // M_ln bf16 [32768][256]      = 16 MB
#define WS_WQKVG  ((size_t)(2097152+16777216))      // fused weights bf16 [1024][256]
#define WS_WO     (WS_WQKVG + (size_t)524288)       // W_o bf16 [256][256]
#define WS_QKVG   (WS_WO + (size_t)131072)          // qkv+gate bf16 [32768][1024] = 64 MB
#define WS_WA     (WS_QKVG + (size_t)67108864)      // attn out bf16 [32768][256]  = 16 MB

static __device__ __forceinline__ v16bf bfcat(v8bf lo, v8bf hi) {
  return __builtin_shufflevector(lo, hi, 0,1,2,3,4,5,6,7,8,9,10,11,12,13,14,15);
}
static __device__ __forceinline__ v8f zero8() {
  v8f z = {0.f,0.f,0.f,0.f,0.f,0.f,0.f,0.f};
  return z;
}
static __device__ __forceinline__ v8f wmma_bf16(v16bf a, v16bf b, v8f c) {
  // D = A(16x32 bf16) * B(32x16 bf16) + C(16x16 f32)
  return __builtin_amdgcn_wmma_f32_16x16x32_bf16(false, a, false, b, (short)0, c, false, false);
}
// Async 16B global->LDS copy, tracked by ASYNCcnt (GLOBAL_LOAD_ASYNC_TO_LDS_B128).
// LDS byte offset = low 32 bits of the generic pointer (ISA 10.2 aperture rule).
static __device__ __forceinline__ void async_copy16(void* lds_dst, const void* gsrc) {
  const unsigned lds_off = (unsigned)(unsigned long long)lds_dst;
  const unsigned long long ga = (unsigned long long)gsrc;
  asm volatile("global_load_async_to_lds_b128 %0, %1, off"
               :: "v"(lds_off), "v"(ga) : "memory");
}
static __device__ __forceinline__ void wait_async0() {
  asm volatile("s_wait_asynccnt 0x0" ::: "memory");
}
static __device__ __forceinline__ float redsum32(float v) {
#pragma unroll
  for (int m = 16; m >= 1; m >>= 1) v += __shfl_xor(v, m, 32);
  return v;
}
static __device__ __forceinline__ float redsum16(float v) {
#pragma unroll
  for (int m = 8; m >= 1; m >>= 1) v += __shfl_xor(v, m, 16);
  return v;
}
static __device__ __forceinline__ float redmax16(float v) {
#pragma unroll
  for (int m = 8; m >= 1; m >>= 1) v = fmaxf(v, __shfl_xor(v, m, 16));
  return v;
}

// ---------------------------------------------------------------------------
// 1) LayerNorm over D_NODE=256, output bf16.  1 wave per row, 8 rows / block.
// ---------------------------------------------------------------------------
__global__ __launch_bounds__(256) void ln_m_kernel(
    const float* __restrict__ M_raw, const float* __restrict__ scale,
    const float* __restrict__ bias, bf16_t* __restrict__ Mln) {
  const int wave = threadIdx.x >> 5, lane = threadIdx.x & 31;
  const int row = blockIdx.x * 8 + wave;              // < 32768
  const float* x = M_raw + (size_t)row * DNODE + lane * 8;
  float4 a = *(const float4*)x;
  float4 b = *(const float4*)(x + 4);
  float v[8] = {a.x, a.y, a.z, a.w, b.x, b.y, b.z, b.w};
  float s = 0.f, sq = 0.f;
#pragma unroll
  for (int j = 0; j < 8; ++j) { s += v[j]; sq += v[j] * v[j]; }
  s = redsum32(s); sq = redsum32(sq);
  const float mu = s * (1.f / DNODE);
  const float var = sq * (1.f / DNODE) - mu * mu;
  const float rinv = rsqrtf(var + 1e-5f);
  v8bf o;
#pragma unroll
  for (int j = 0; j < 8; ++j) {
    const int c = lane * 8 + j;
    o[j] = (bf16_t)((v[j] - mu) * rinv * scale[c] + bias[c]);
  }
  *(v8bf*)(Mln + (size_t)row * DNODE + lane * 8) = o;
}

// ---------------------------------------------------------------------------
// 2) LayerNorm(Z) over D_PAIR=128 fused with pair_bias = Zn . W_b^T.
//    1 wave per (q,k) row; writes pair_bias[h][q][k] f32.
// ---------------------------------------------------------------------------
__global__ __launch_bounds__(256) void ln_z_pairbias_kernel(
    const float* __restrict__ Z, const float* __restrict__ scale,
    const float* __restrict__ bias, const float* __restrict__ W_b,
    float* __restrict__ pb) {
  const int wave = threadIdx.x >> 5, lane = threadIdx.x & 31;
  const int row = blockIdx.x * 8 + wave;              // < 65536 (q*256 + k)
  const float4 z = *(const float4*)(Z + (size_t)row * DPAIR + lane * 4);
  float v[4] = {z.x, z.y, z.z, z.w};
  float s = 0.f, sq = 0.f;
#pragma unroll
  for (int j = 0; j < 4; ++j) { s += v[j]; sq += v[j] * v[j]; }
  s = redsum32(s); sq = redsum32(sq);
  const float mu = s * (1.f / DPAIR);
  const float var = sq * (1.f / DPAIR) - mu * mu;
  const float rinv = rsqrtf(var + 1e-5f);
  float zn[4];
#pragma unroll
  for (int j = 0; j < 4; ++j) {
    const int c = lane * 4 + j;
    zn[j] = (v[j] - mu) * rinv * scale[c] + bias[c];
  }
#pragma unroll
  for (int h = 0; h < NHEAD; ++h) {
    const float4 w = *(const float4*)(W_b + h * DPAIR + lane * 4);
    float p = zn[0] * w.x + zn[1] * w.y + zn[2] * w.z + zn[3] * w.w;
    p = redsum32(p);
    if (lane == 0) pb[(size_t)h * 65536 + row] = p;
  }
}

// ---------------------------------------------------------------------------
// 3) Weight conversion: fused [W_qkv; W_gate] -> bf16 [1024][256], W_o -> bf16.
// ---------------------------------------------------------------------------
__global__ __launch_bounds__(256) void convert_weights_kernel(
    const float* __restrict__ W_qkv, const float* __restrict__ W_gate,
    const float* __restrict__ W_o, bf16_t* __restrict__ wqkvg,
    bf16_t* __restrict__ wo) {
  const int total = EDIM * DNODE + DNODE * DNODE;     // 262144 + 65536
  for (int i = blockIdx.x * blockDim.x + threadIdx.x; i < total;
       i += gridDim.x * blockDim.x) {
    if (i < EDIM * DNODE) {
      const int r = i >> 8, c = i & 255;
      const float v = (r < OFF_G) ? W_qkv[i] : W_gate[(r - OFF_G) * DNODE + c];
      wqkvg[i] = (bf16_t)v;
    } else {
      const int j = i - EDIM * DNODE;
      wo[j] = (bf16_t)W_o[j];
    }
  }
}

// ---------------------------------------------------------------------------
// 4) WMMA GEMM: QKVG[32768,1024] = Mln[32768,256] x Wqkvg[1024,256]^T (bf16).
//    Block: 8 waves, 128 rows x 64 cols; weight panel async-staged into LDS.
// ---------------------------------------------------------------------------
__global__ __launch_bounds__(256) void gemm_qkvg_kernel(
    const bf16_t* __restrict__ A, const bf16_t* __restrict__ Bw,
    bf16_t* __restrict__ C) {
  __shared__ bf16_t ldsB[64 * 256];                   // 32 KB
  {
    const bf16_t* src = Bw + (size_t)blockIdx.y * 64 * 256;
    for (int i = threadIdx.x; i < (64 * 256) / 8; i += 256)
      async_copy16(ldsB + i * 8, src + i * 8);
    wait_async0();
  }
  __syncthreads();
  const int wave = threadIdx.x >> 5, lane = threadIdx.x & 31;
  const int n = lane & 15, hq = lane >> 4;
  const int m0 = blockIdx.x * 128 + wave * 16;
  v8f acc[4] = {zero8(), zero8(), zero8(), zero8()};
  const bf16_t* arow = A + (size_t)(m0 + n) * 256;
#pragma unroll
  for (int k0 = 0; k0 < 256; k0 += 32) {
    const v16bf a = bfcat(*(const v8bf*)(arow + k0 + hq * 16),
                          *(const v8bf*)(arow + k0 + hq * 16 + 8));
#pragma unroll
    for (int nt = 0; nt < 4; ++nt) {
      const bf16_t* brow = ldsB + (nt * 16 + n) * 256 + k0 + hq * 16;
      const v16bf b = bfcat(*(const v8bf*)brow, *(const v8bf*)(brow + 8));
      acc[nt] = wmma_bf16(a, b, acc[nt]);
    }
  }
#pragma unroll
  for (int nt = 0; nt < 4; ++nt)
#pragma unroll
    for (int r = 0; r < 8; ++r) {
      const int row = m0 + r + 8 * hq;
      const int col = blockIdx.y * 64 + nt * 16 + n;
      C[(size_t)row * EDIM + col] = (bf16_t)acc[nt][r];
    }
}

// ---------------------------------------------------------------------------
// 5) Flash attention per (s, h, 16-row q tile). 1 wave per block.
//    logits = wmma(Q,K^T)*scale + pair_bias + mask_bias; online softmax;
//    P transposed via LDS into A-layout; O += wmma(P, V).
//    V tile is async-copied to LDS and waited only before the PV wmma.
// ---------------------------------------------------------------------------
__global__ __launch_bounds__(32) void attn_kernel(
    const bf16_t* __restrict__ qkvg, const float* __restrict__ pb,
    const float* __restrict__ mask, bf16_t* __restrict__ wa) {
  __shared__ bf16_t ldsV[32][32];                     // V tile (k-local x c)
  __shared__ bf16_t ldsP[16][32];                     // P tile (q-row x k-local)
  const int bid = blockIdx.x;
  const int qt = bid & 15, h = (bid >> 4) & 7, s = bid >> 7;
  const int lane = threadIdx.x;
  const int n = lane & 15, hq = lane >> 4;

  // Q A-fragment (row = lane&15, K split by half)
  const bf16_t* qp = qkvg + (size_t)(s * N_RES + qt * 16 + n) * EDIM + h * CDIM;
  const v16bf aq = bfcat(*(const v8bf*)(qp + hq * 16),
                         *(const v8bf*)(qp + hq * 16 + 8));

  float mstate[8], lstate[8];
  v8f o0 = zero8(), o1 = zero8();
#pragma unroll
  for (int r = 0; r < 8; ++r) { mstate[r] = -1e30f; lstate[r] = 0.f; }

#pragma unroll 1
  for (int kb = 0; kb < 8; ++kb) {
    const int k0 = kb * 32;
    __syncthreads();   // previous iteration's LDS reads done
    // async-stage V tile: each lane copies one 32-element row (4 x 16B)
    const bf16_t* vp =
        qkvg + (size_t)(s * N_RES + k0 + lane) * EDIM + OFF_V + h * CDIM;
    async_copy16(&ldsV[lane][0],  vp);
    async_copy16(&ldsV[lane][8],  vp + 8);
    async_copy16(&ldsV[lane][16], vp + 16);
    async_copy16(&ldsV[lane][24], vp + 24);
    // K B-fragments (col = lane&15 -> k row, K-dim = c split by half)
    const bf16_t* kp0 =
        qkvg + (size_t)(s * N_RES + k0 + n) * EDIM + OFF_K + h * CDIM + hq * 16;
    const bf16_t* kp1 = kp0 + (size_t)16 * EDIM;
    const v16bf bk0 = bfcat(*(const v8bf*)kp0, *(const v8bf*)(kp0 + 8));
    const v16bf bk1 = bfcat(*(const v8bf*)kp1, *(const v8bf*)(kp1 + 8));
    if (kb < 7) {                                     // prefetch next tiles
      __builtin_prefetch(kp0 + (size_t)32 * EDIM, 0, 0);
      __builtin_prefetch(vp + (size_t)32 * EDIM, 0, 0);
    }
    v8f lg0 = wmma_bf16(aq, bk0, zero8());
    v8f lg1 = wmma_bf16(aq, bk1, zero8());

    const float mb0 = 1e9f * (mask[s * N_RES + k0 + n] - 1.f);
    const float mb1 = 1e9f * (mask[s * N_RES + k0 + 16 + n] - 1.f);
    float l0[8], l1[8];
#pragma unroll
    for (int r = 0; r < 8; ++r) {
      const int qr = qt * 16 + r + 8 * hq;
      const float* pbrow = pb + ((size_t)(h * N_RES + qr)) * N_RES + k0;
      l0[r] = lg0[r] * QK_SCALE + pbrow[n] + mb0;
      l1[r] = lg1[r] * QK_SCALE + pbrow[16 + n] + mb1;
    }
#pragma unroll
    for (int r = 0; r < 8; ++r) {
      const float rm = redmax16(fmaxf(l0[r], l1[r]));
      const float mn = fmaxf(mstate[r], rm);
      const float alpha = __expf(mstate[r] - mn);
      mstate[r] = mn;
      const float p0 = __expf(l0[r] - mn);
      const float p1 = __expf(l1[r] - mn);
      lstate[r] = lstate[r] * alpha + redsum16(p0 + p1);
      o0[r] *= alpha;
      o1[r] *= alpha;
      ldsP[r + 8 * hq][n] = (bf16_t)p0;
      ldsP[r + 8 * hq][16 + n] = (bf16_t)p1;
    }
    wait_async0();     // V tile resident in LDS
    __syncthreads();
    // P as A-fragment, V as B-fragments from LDS
    const v16bf pa = bfcat(*(const v8bf*)&ldsP[n][hq * 16],
                           *(const v8bf*)&ldsP[n][hq * 16 + 8]);
    v16bf bv0, bv1;
#pragma unroll
    for (int j = 0; j < 16; ++j) {
      bv0[j] = ldsV[hq * 16 + j][n];
      bv1[j] = ldsV[hq * 16 + j][16 + n];
    }
    o0 = wmma_bf16(pa, bv0, o0);
    o1 = wmma_bf16(pa, bv1, o1);
  }
  // normalize and write (un-gated) attention output
#pragma unroll
  for (int r = 0; r < 8; ++r) {
    const float inv = 1.f / lstate[r];
    const int qr = qt * 16 + r + 8 * hq;
    bf16_t* wrow = wa + (size_t)(s * N_RES + qr) * DNODE + h * CDIM;
    wrow[n] = (bf16_t)(o0[r] * inv);
    wrow[16 + n] = (bf16_t)(o1[r] * inv);
  }
}

// ---------------------------------------------------------------------------
// 6) Apply sigmoid gate (from QKVG cols 768..1023) to attention output.
// ---------------------------------------------------------------------------
__global__ __launch_bounds__(256) void gate_kernel(
    const bf16_t* __restrict__ qkvg, const float* __restrict__ gbias,
    bf16_t* __restrict__ wa) {
  const int total = 32768 * 256;
  for (int i = blockIdx.x * blockDim.x + threadIdx.x; i < total;
       i += gridDim.x * blockDim.x) {
    const int row = i >> 8, e = i & 255;
    const float g = (float)qkvg[(size_t)row * EDIM + OFF_G + e] + gbias[e];
    const float sg = 1.f / (1.f + __expf(-g));
    wa[i] = (bf16_t)((float)wa[i] * sg);
  }
}

// ---------------------------------------------------------------------------
// 7) WMMA GEMM + epilogue: out = WA x W_o^T + M_raw + out_bias (f32 out).
// ---------------------------------------------------------------------------
__global__ __launch_bounds__(256) void gemm_out_kernel(
    const bf16_t* __restrict__ A, const bf16_t* __restrict__ Bw,
    const float* __restrict__ M_raw, const float* __restrict__ out_bias,
    float* __restrict__ Out) {
  __shared__ bf16_t ldsB[64 * 256];                   // 32 KB
  {
    const bf16_t* src = Bw + (size_t)blockIdx.y * 64 * 256;
    for (int i = threadIdx.x; i < (64 * 256) / 8; i += 256)
      async_copy16(ldsB + i * 8, src + i * 8);
    wait_async0();
  }
  __syncthreads();
  const int wave = threadIdx.x >> 5, lane = threadIdx.x & 31;
  const int n = lane & 15, hq = lane >> 4;
  const int m0 = blockIdx.x * 128 + wave * 16;
  v8f acc[4] = {zero8(), zero8(), zero8(), zero8()};
  const bf16_t* arow = A + (size_t)(m0 + n) * 256;
#pragma unroll
  for (int k0 = 0; k0 < 256; k0 += 32) {
    const v16bf a = bfcat(*(const v8bf*)(arow + k0 + hq * 16),
                          *(const v8bf*)(arow + k0 + hq * 16 + 8));
#pragma unroll
    for (int nt = 0; nt < 4; ++nt) {
      const bf16_t* brow = ldsB + (nt * 16 + n) * 256 + k0 + hq * 16;
      const v16bf b = bfcat(*(const v8bf*)brow, *(const v8bf*)(brow + 8));
      acc[nt] = wmma_bf16(a, b, acc[nt]);
    }
  }
#pragma unroll
  for (int nt = 0; nt < 4; ++nt)
#pragma unroll
    for (int r = 0; r < 8; ++r) {
      const int row = m0 + r + 8 * hq;
      const int col = blockIdx.y * 64 + nt * 16 + n;
      Out[(size_t)row * DNODE + col] =
          acc[nt][r] + M_raw[(size_t)row * DNODE + col] + out_bias[col];
    }
}

// ---------------------------------------------------------------------------
extern "C" void kernel_launch(void* const* d_in, const int* in_sizes, int n_in,
                              void* d_out, int out_size, void* d_ws,
                              size_t ws_size, hipStream_t stream) {
  const float* M_raw      = (const float*)d_in[0];
  const float* Z          = (const float*)d_in[1];
  const float* M_mask     = (const float*)d_in[2];
  const float* ln_m_scale = (const float*)d_in[3];
  const float* ln_m_bias  = (const float*)d_in[4];
  const float* ln_z_scale = (const float*)d_in[5];
  const float* ln_z_bias  = (const float*)d_in[6];
  const float* W_b        = (const float*)d_in[7];
  const float* W_qkv      = (const float*)d_in[8];
  const float* W_gate     = (const float*)d_in[9];
  const float* gbias      = (const float*)d_in[10];
  const float* W_o        = (const float*)d_in[11];
  const float* out_bias   = (const float*)d_in[12];
  float* out = (float*)d_out;

  char* ws = (char*)d_ws;
  float*  pb    = (float*)(ws + WS_PB);
  bf16_t* Mln   = (bf16_t*)(ws + WS_MLN);
  bf16_t* wqkvg = (bf16_t*)(ws + WS_WQKVG);
  bf16_t* wo    = (bf16_t*)(ws + WS_WO);
  bf16_t* qkvg  = (bf16_t*)(ws + WS_QKVG);
  bf16_t* wa    = (bf16_t*)(ws + WS_WA);

  convert_weights_kernel<<<1280, 256, 0, stream>>>(W_qkv, W_gate, W_o, wqkvg, wo);
  ln_m_kernel<<<4096, 256, 0, stream>>>(M_raw, ln_m_scale, ln_m_bias, Mln);
  ln_z_pairbias_kernel<<<8192, 256, 0, stream>>>(Z, ln_z_scale, ln_z_bias, W_b, pb);
  gemm_qkvg_kernel<<<dim3(256, 16), 256, 0, stream>>>(Mln, wqkvg, qkvg);
  attn_kernel<<<16384, 32, 0, stream>>>(qkvg, pb, M_mask, wa);
  gate_kernel<<<8192, 256, 0, stream>>>(qkvg, gbias, wa);
  gemm_out_kernel<<<dim3(256, 4), 256, 0, stream>>>(wa, wo, M_raw, out_bias, out);
}
